// MultiHeadAttentionWithRoPE_24653112279153
// MI455X (gfx1250) — matrix operand order repaired and checked by reading, not compiled, not run on hardware
//
#include <hip/hip_runtime.h>

// ---------- types matching CDNA5 WMMA builtins ----------
typedef __attribute__((ext_vector_type(16))) __bf16 v16bf;
typedef __attribute__((ext_vector_type(8)))  float  v8f;

union V16 {
  v16bf v;
  unsigned short u[16];
  uint4 q[2];
};

__device__ __forceinline__ unsigned short f2bf(float f) {
  unsigned int u = __float_as_uint(f);
  u = (u + 0x7FFFu + ((u >> 16) & 1u)) >> 16;  // round-to-nearest-even
  return (unsigned short)u;
}

// ---------- fp32 -> bf16 convert ----------
__global__ void cvt_f32_bf16_kernel(const float* __restrict__ src,
                                    unsigned short* __restrict__ dst, int n) {
  int i = blockIdx.x * blockDim.x + threadIdx.x;
  int stride = gridDim.x * blockDim.x;
  for (; i < n; i += stride) dst[i] = f2bf(src[i]);
}

// ---------- bf16 WMMA GEMM with async global->LDS double buffering ----------
// out[M,N] = A[M,K] * W[N,K]^T (+bias), optional fp32/bf16 out, optional row permute.
// Workgroup: 128 threads = 4 waves (2x2), tile 128x128. Each wave: 64x64 (16 WMMA accums).
// K stepped by 32; A/B k-slabs (128x32 bf16 = 8KB each) staged in LDS via
// GLOBAL_LOAD_ASYNC_TO_LDS_B128 (ASYNCcnt), double buffered.
__global__ __launch_bounds__(128)
void gemm_bf16_wmma_kernel(const unsigned short* __restrict__ A,
                           const unsigned short* __restrict__ W,
                           const float* __restrict__ bias,
                           float* __restrict__ outF,
                           unsigned short* __restrict__ outB,
                           int M, int N, int K, int permDiv, int permMul)
{
  // LDS: A buffers [2][128][32] bf16, then B buffers [2][128][32] bf16  (32 KB)
  __shared__ __align__(16) unsigned short smem[4 * 4096];
  const unsigned smbase = (unsigned)(uintptr_t)(void*)&smem[0];  // LDS byte offset

  const int t    = threadIdx.x;           // 0..127
  const int lane = t & 31;
  const int wave = t >> 5;                // 0..3
  const int wy   = wave >> 1;             // wave row (0..1)
  const int wx   = wave & 1;              // wave col (0..1)
  const int ln   = lane & 15;
  const int lh   = lane >> 4;
  const int kb   = lh * 8;

  const int m0  = blockIdx.x * 128;
  const int nb0 = blockIdx.y * 128;

  const unsigned short* Aba = A + (size_t)m0 * K;   // uniform SGPR base
  const unsigned short* Bba = W + (size_t)nb0 * K;  // uniform SGPR base

  const unsigned Kb = (unsigned)K * 2u;             // row stride in bytes
  const unsigned rowA = (unsigned)(t >> 2);         // 0..31 (+32*i)
  const unsigned colC = (unsigned)(t & 3) * 16u;    // 16B chunk within 64B row

  // stage one 128x32 k-slab of A and B into LDS buffer `bufSel`
  auto stage = [&](int bufSel, int k) {
    const unsigned kByte = (unsigned)k * 2u;
    #pragma unroll
    for (int i = 0; i < 4; ++i) {
      unsigned voff = (rowA + 32u * i) * Kb + colC + kByte;
      unsigned ldsA = smbase + (unsigned)bufSel * 8192u + ((unsigned)t + 128u * i) * 16u;
      asm volatile("global_load_async_to_lds_b128 %0, %1, %2"
                   :: "v"(ldsA), "v"(voff), "s"(Aba) : "memory");
      unsigned ldsB = smbase + 16384u + (unsigned)bufSel * 8192u + ((unsigned)t + 128u * i) * 16u;
      asm volatile("global_load_async_to_lds_b128 %0, %1, %2"
                   :: "v"(ldsB), "v"(voff), "s"(Bba) : "memory");
    }
  };

  v8f acc[4][4];
  #pragma unroll
  for (int i = 0; i < 4; ++i)
    #pragma unroll
    for (int j = 0; j < 4; ++j) acc[i][j] = (v8f){};

  const int nk = K >> 5;
  stage(0, 0);
  asm volatile("s_wait_asynccnt 0x0" ::: "memory");
  __syncthreads();

  for (int ks = 0; ks < nk; ++ks) {
    const int cur = ks & 1;
    if (ks + 1 < nk) stage(cur ^ 1, (ks + 1) << 5);

    const unsigned short* As = smem + cur * 4096;          // [128][32]
    const unsigned short* Bs = smem + 8192 + cur * 4096;   // [128][32]

    V16 a[4];
    #pragma unroll
    for (int tm = 0; tm < 4; ++tm) {
      const unsigned short* ap = As + (wy * 64 + tm * 16 + ln) * 32;
      a[tm].q[0] = *(const uint4*)(ap + kb);
      a[tm].q[1] = *(const uint4*)(ap + 16 + kb);
    }
    #pragma unroll
    for (int tn = 0; tn < 4; ++tn) {
      V16 b;
      const unsigned short* bp = Bs + (wx * 64 + tn * 16 + ln) * 32 + kb;
      b.q[0] = *(const uint4*)(bp);
      b.q[1] = *(const uint4*)(bp + 16);
      #pragma unroll
      for (int tm = 0; tm < 4; ++tm)
        acc[tm][tn] = __builtin_amdgcn_wmma_f32_16x16x32_bf16(
            false, a[tm].v, false, b.v, (short)0, acc[tm][tn], false, false);
    }

    asm volatile("s_wait_asynccnt 0x0" ::: "memory");
    __syncthreads();
  }

  #pragma unroll
  for (int tm = 0; tm < 4; ++tm)
    #pragma unroll
    for (int tn = 0; tn < 4; ++tn)
      #pragma unroll
      for (int r = 0; r < 8; ++r) {
        int row = m0 + wy * 64 + tm * 16 + r + 8 * lh;
        int col = nb0 + wx * 64 + tn * 16 + ln;
        if (row >= M || col >= N) continue;
        float v = acc[tm][tn][r];
        if (bias) v += bias[col];
        size_t orow = (size_t)row;
        if (permDiv) {  // row = b*5280 + i1*permDiv + i2 -> orow = b*5280 + i2*permMul + i1
          int b = row / 5280;
          int rem = row - b * 5280;
          int i1 = rem / permDiv;
          int i2 = rem - i1 * permDiv;
          orow = (size_t)b * 5280 + (size_t)i2 * permMul + i1;
        }
        if (outF) outF[orow * N + col] = v;
        if (outB) outB[orow * N + col] = f2bf(v);
      }
}

// ---------- fused RMSNorm + RoPE + flash attention per (batch, head) ----------
#define SPAD 224
__global__ __launch_bounds__(256)
void attn_rope_kernel(const float* __restrict__ qkv,      // [NB*S, 3072]
                      unsigned short* __restrict__ outB,  // [NB*S, 1024] bf16
                      const float* __restrict__ qg,
                      const float* __restrict__ kg,
                      int S)
{
  __shared__ __align__(16) unsigned short ks[SPAD * 64];    // normed+roped K, row-major
  __shared__ __align__(16) unsigned short vT[64 * SPAD];    // V transposed [d][key]
  __shared__ __align__(16) unsigned short pst[8 * 16 * 32]; // per-wave P staging

  const int n = blockIdx.x;
  const int h = blockIdx.y;
  const int tid = threadIdx.x;
  const size_t base = (size_t)n * S * 3072;
  const int cq = h * 64, ck = 1024 + h * 64, cv = 2048 + h * 64;
  const float LOG1E4 = 9.2103403719761836f;

  // ---- stage K (rmsnorm + rope) and V^T into LDS ----
  for (int r = tid; r < SPAD; r += 256) {
    if (r < S) {
      const float* kr = qkv + base + (size_t)r * 3072 + ck;
      const float* vr = qkv + base + (size_t)r * 3072 + cv;
      float t[64]; float ss = 0.f;
      #pragma unroll
      for (int d = 0; d < 64; ++d) { float x = kr[d]; t[d] = x; ss += x * x; }
      float inv = rsqrtf(ss * (1.0f / 64.0f) + 1e-6f);
      #pragma unroll
      for (int i = 0; i < 32; ++i) {
        float theta = __expf(-(float)(2 * i) * (1.0f / 64.0f) * LOG1E4);
        float ang = (float)r * theta;
        float sn = __sinf(ang), cs = __cosf(ang);
        float x0 = t[2 * i]     * inv * kg[2 * i];
        float x1 = t[2 * i + 1] * inv * kg[2 * i + 1];
        ks[r * 64 + 2 * i]     = f2bf(x0 * cs - x1 * sn);
        ks[r * 64 + 2 * i + 1] = f2bf(x1 * cs + x0 * sn);
      }
      #pragma unroll
      for (int d = 0; d < 64; ++d) vT[d * SPAD + r] = f2bf(vr[d]);
    } else {
      for (int d = 0; d < 64; ++d) { ks[r * 64 + d] = 0; vT[d * SPAD + r] = 0; }
    }
  }
  __syncthreads();

  const int wv = tid >> 5;
  const int lane = tid & 31;
  const int ln = lane & 15;
  const int lh = lane >> 4;
  const int kb = lh * 8;
  unsigned short* myp = pst + wv * 512;
  const int nqc = (S + 15) >> 4;
  const int nkb = (S + 31) >> 5;
  const float scale = 0.125f;  // 1/sqrt(64)

  for (int qc = wv; qc < nqc; qc += 8) {
    int qrow = qc * 16 + ln;
    int qr = qrow < S ? qrow : (S - 1);
    const float* qp = qkv + base + (size_t)qr * 3072 + cq;

    // load Q row into WMMA-A layout with rmsnorm + rope (rope pairs stay lane-local)
    float qv[32]; float ss = 0.f; int idx = 0;
    #pragma unroll
    for (int t = 0; t < 2; ++t)
      #pragma unroll
      for (int hf = 0; hf < 2; ++hf)
        #pragma unroll
        for (int j = 0; j < 8; ++j) {
          int d = t * 32 + hf * 16 + kb + j;
          float x = qp[d];
          qv[idx++] = x; ss += x * x;
        }
    ss += __shfl_xor(ss, 16, 32);  // lane and lane^16 hold complementary halves of the row
    float inv = rsqrtf(ss * (1.0f / 64.0f) + 1e-6f);

    V16 a0, a1;
    idx = 0;
    #pragma unroll
    for (int t = 0; t < 2; ++t)
      #pragma unroll
      for (int hf = 0; hf < 2; ++hf)
        #pragma unroll
        for (int j = 0; j < 8; j += 2) {
          int d = t * 32 + hf * 16 + kb + j;  // even
          float theta = __expf(-(float)d * (1.0f / 64.0f) * LOG1E4);
          float ang = (float)qrow * theta;
          float sn = __sinf(ang), cs = __cosf(ang);
          float x0 = qv[idx]     * inv * qg[d];
          float x1 = qv[idx + 1] * inv * qg[d + 1];
          unsigned short e0 = f2bf(x0 * cs - x1 * sn);
          unsigned short e1 = f2bf(x1 * cs + x0 * sn);
          if (t == 0) { a0.u[hf * 8 + j] = e0; a0.u[hf * 8 + j + 1] = e1; }
          else        { a1.u[hf * 8 + j] = e0; a1.u[hf * 8 + j + 1] = e1; }
          idx += 2;
        }

    float m[8], lsum[8];
    v8f o0 = {}, o1 = {}, o2 = {}, o3 = {};
    #pragma unroll
    for (int r = 0; r < 8; ++r) { m[r] = -1e30f; lsum[r] = 0.f; }

    for (int kt = 0; kt < nkb; ++kt) {
      // scores: two 16x16 tiles over keys kt*32..kt*32+31, K-dim = d (64 -> 2 WMMAs each)
      v8f s0 = {}, s1 = {};
      {
        V16 b;
        const unsigned short* kp = ks + (size_t)(kt * 32 + ln) * 64 + kb;
        b.q[0] = *(const uint4*)(kp);      b.q[1] = *(const uint4*)(kp + 16);
        s0 = __builtin_amdgcn_wmma_f32_16x16x32_bf16(false, a0.v, false, b.v, (short)0, s0, false, false);
        b.q[0] = *(const uint4*)(kp + 32); b.q[1] = *(const uint4*)(kp + 48);
        s0 = __builtin_amdgcn_wmma_f32_16x16x32_bf16(false, a1.v, false, b.v, (short)0, s0, false, false);
        kp = ks + (size_t)(kt * 32 + 16 + ln) * 64 + kb;
        b.q[0] = *(const uint4*)(kp);      b.q[1] = *(const uint4*)(kp + 16);
        s1 = __builtin_amdgcn_wmma_f32_16x16x32_bf16(false, a0.v, false, b.v, (short)0, s1, false, false);
        b.q[0] = *(const uint4*)(kp + 32); b.q[1] = *(const uint4*)(kp + 48);
        s1 = __builtin_amdgcn_wmma_f32_16x16x32_bf16(false, a1.v, false, b.v, (short)0, s1, false, false);
      }
      float km0 = (kt * 32 + ln)      < S ? 0.f : -1e30f;
      float km1 = (kt * 32 + 16 + ln) < S ? 0.f : -1e30f;
      float cf[8];
      #pragma unroll
      for (int r = 0; r < 8; ++r) {
        float x0 = s0[r] * scale + km0;
        float x1 = s1[r] * scale + km1;
        float mx = fmaxf(x0, x1);
        mx = fmaxf(mx, __shfl_xor(mx, 1, 32));
        mx = fmaxf(mx, __shfl_xor(mx, 2, 32));
        mx = fmaxf(mx, __shfl_xor(mx, 4, 32));
        mx = fmaxf(mx, __shfl_xor(mx, 8, 32));
        float mn = fmaxf(m[r], mx);
        float c = __expf(m[r] - mn);
        x0 = __expf(x0 - mn);
        x1 = __expf(x1 - mn);
        float rs = x0 + x1;
        rs += __shfl_xor(rs, 1, 32);
        rs += __shfl_xor(rs, 2, 32);
        rs += __shfl_xor(rs, 4, 32);
        rs += __shfl_xor(rs, 8, 32);
        lsum[r] = lsum[r] * c + rs;
        m[r] = mn;
        cf[r] = c;
        // stage P (C-layout -> row-major 16x32) for re-load in A-layout
        myp[(r + 8 * lh) * 32 + ln]      = f2bf(x0);
        myp[(r + 8 * lh) * 32 + 16 + ln] = f2bf(x1);
      }
      #pragma unroll
      for (int r = 0; r < 8; ++r) { o0[r] *= cf[r]; o1[r] *= cf[r]; o2[r] *= cf[r]; o3[r] *= cf[r]; }

      V16 pa;
      pa.q[0] = *(const uint4*)(myp + ln * 32 + kb);
      pa.q[1] = *(const uint4*)(myp + ln * 32 + 16 + kb);
      {
        V16 bv;
        const unsigned short* vp = vT + (size_t)(ln) * SPAD + kt * 32 + kb;
        bv.q[0] = *(const uint4*)vp; bv.q[1] = *(const uint4*)(vp + 16);
        o0 = __builtin_amdgcn_wmma_f32_16x16x32_bf16(false, pa.v, false, bv.v, (short)0, o0, false, false);
        vp = vT + (size_t)(16 + ln) * SPAD + kt * 32 + kb;
        bv.q[0] = *(const uint4*)vp; bv.q[1] = *(const uint4*)(vp + 16);
        o1 = __builtin_amdgcn_wmma_f32_16x16x32_bf16(false, pa.v, false, bv.v, (short)0, o1, false, false);
        vp = vT + (size_t)(32 + ln) * SPAD + kt * 32 + kb;
        bv.q[0] = *(const uint4*)vp; bv.q[1] = *(const uint4*)(vp + 16);
        o2 = __builtin_amdgcn_wmma_f32_16x16x32_bf16(false, pa.v, false, bv.v, (short)0, o2, false, false);
        vp = vT + (size_t)(48 + ln) * SPAD + kt * 32 + kb;
        bv.q[0] = *(const uint4*)vp; bv.q[1] = *(const uint4*)(vp + 16);
        o3 = __builtin_amdgcn_wmma_f32_16x16x32_bf16(false, pa.v, false, bv.v, (short)0, o3, false, false);
      }
    }

    #pragma unroll
    for (int r = 0; r < 8; ++r) {
      int q = qc * 16 + r + 8 * lh;
      if (q >= S) continue;
      float invl = 1.0f / lsum[r];
      size_t ob = ((size_t)n * S + q) * 1024 + h * 64;
      outB[ob + ln]      = f2bf(o0[r] * invl);
      outB[ob + 16 + ln] = f2bf(o1[r] * invl);
      outB[ob + 32 + ln] = f2bf(o2[r] * invl);
      outB[ob + 48 + ln] = f2bf(o3[r] * invl);
    }
  }
}

// ---------- host orchestration ----------
extern "C" void kernel_launch(void* const* d_in, const int* in_sizes, int n_in,
                              void* d_out, int out_size, void* d_ws, size_t ws_size,
                              hipStream_t stream)
{
  const float* x       = (const float*)d_in[0];
  const float* w_sqkv  = (const float*)d_in[1];
  const float* w_sproj = (const float*)d_in[2];
  const float* b_sproj = (const float*)d_in[3];
  const float* w_tqkv  = (const float*)d_in[4];
  const float* w_tproj = (const float*)d_in[5];
  const float* b_tproj = (const float*)d_in[6];
  const float* sq_g    = (const float*)d_in[7];
  const float* sk_g    = (const float*)d_in[8];
  const float* tq_g    = (const float*)d_in[9];
  const float* tk_g    = (const float*)d_in[10];

  const int E = 1024, E3 = 3072;
  const int M = 4 * 5280;  // 21120 tokens (same for both blocks)

  char* ws = (char*)d_ws;
  size_t off = 0;
  auto alloc = [&](size_t bytes) -> char* {
    char* p = ws + off;
    off += (bytes + 255) & ~(size_t)255;
    return p;
  };
  unsigned short* xbf     = (unsigned short*)alloc((size_t)M * E * 2);
  unsigned short* wqkv_s  = (unsigned short*)alloc((size_t)E3 * E * 2);
  unsigned short* wproj_s = (unsigned short*)alloc((size_t)E * E * 2);
  unsigned short* wqkv_t  = (unsigned short*)alloc((size_t)E3 * E * 2);
  unsigned short* wproj_t = (unsigned short*)alloc((size_t)E * E * 2);
  float*          qkv     = (float*)alloc((size_t)M * E3 * 4);
  unsigned short* attnO   = (unsigned short*)alloc((size_t)M * E * 2);
  unsigned short* xt      = (unsigned short*)alloc((size_t)M * E * 2);
  if (off > ws_size) return;

  dim3 blk256(256);
  cvt_f32_bf16_kernel<<<dim3(2048), blk256, 0, stream>>>(x, xbf, M * E);
  cvt_f32_bf16_kernel<<<dim3(1024), blk256, 0, stream>>>(w_sqkv, wqkv_s, E3 * E);
  cvt_f32_bf16_kernel<<<dim3(512),  blk256, 0, stream>>>(w_sproj, wproj_s, E * E);
  cvt_f32_bf16_kernel<<<dim3(1024), blk256, 0, stream>>>(w_tqkv, wqkv_t, E3 * E);
  cvt_f32_bf16_kernel<<<dim3(512),  blk256, 0, stream>>>(w_tproj, wproj_t, E * E);

  dim3 gblk(128);
  // ----- spatial block: xs rows are exactly x rows (b*5280 + f*220 + p) -----
  gemm_bf16_wmma_kernel<<<dim3(M / 128, E3 / 128), gblk, 0, stream>>>(
      xbf, wqkv_s, nullptr, qkv, nullptr, M, E3, E, 0, 0);
  attn_rope_kernel<<<dim3(96, 16), blk256, 0, stream>>>(qkv, attnO, sq_g, sk_g, 220);
  // proj + bias, write bf16 permuted (b,f,p) -> (b,p,f) so xt rows = (b*220+p)*24+f
  gemm_bf16_wmma_kernel<<<dim3(M / 128, E / 128), gblk, 0, stream>>>(
      attnO, wproj_s, b_sproj, nullptr, xt, M, E, E, /*permDiv=*/220, /*permMul=*/24);

  // ----- temporal block -----
  gemm_bf16_wmma_kernel<<<dim3(M / 128, E3 / 128), gblk, 0, stream>>>(
      xt, wqkv_t, nullptr, qkv, nullptr, M, E3, E, 0, 0);
  attn_rope_kernel<<<dim3(880, 16), blk256, 0, stream>>>(qkv, attnO, tq_g, tk_g, 24);
  // final proj + bias, fp32 out permuted (b,p,f) -> (b,f,p)
  gemm_bf16_wmma_kernel<<<dim3(M / 128, E / 128), gblk, 0, stream>>>(
      attnO, wproj_t, b_tproj, (float*)d_out, nullptr, M, E, E, /*permDiv=*/24, /*permMul=*/220);
}
